// GCNEncoder_69621419868189
// MI455X (gfx1250) — matrix operand order repaired and checked by reading, not compiled, not run on hardware
//
#include <hip/hip_runtime.h>

typedef __attribute__((ext_vector_type(16))) __bf16 v16bf;
typedef __attribute__((ext_vector_type(8)))  float  v8f;

#define HID 128
#define GRAPHS 128

// ---------------------------------------------------------------------------
// Pack a row-major [K=128, N=128] f32 weight matrix into bf16 B-fragment
// layout: tile t = kt*8+nt holds a 32(K)x16(N) block, 512 bf16; lane l of a
// wave reads 16 contiguous bf16 at t*512 + l*16, covering
//   k = kt*32 + ((l&16)>>1) + (j&7) + ((j&8)<<1)  (j=0..15),  n = nt*16 + (l&15)
// = the CDNA5 16-bit B-matrix VGPR layout (ISA 7.12.2).
// ---------------------------------------------------------------------------
__global__ __launch_bounds__(256) void pack_w_kernel(const float* __restrict__ W,
                                                     __bf16* __restrict__ Wp) {
    int i = blockIdx.x * 256 + threadIdx.x;
    if (i >= HID * HID) return;
    int k  = i >> 7, n = i & 127;
    int kt = k >> 5, nt = n >> 4;
    int kin = k & 31;
    int lane = (n & 15) | ((kin & 8) << 1);
    int j    = (kin & 7) | ((kin & 16) >> 1);
    Wp[(size_t)((kt * 8 + nt) * 512 + lane * 16 + j)] = (__bf16)W[i];
}

// ---------------------------------------------------------------------------
// bf16 WMMA GEMM: out[m, 0:128] = A[idx ? idx[m] : m, 0:128] @ W + bias
// One wave = 16 rows x 128 cols; 4 K-chunks x 8 N-tiles = 32 v_wmma per wave.
// `nclamp`: valid input rows (A row index clamped -> EXEC all-ones for WMMA).
// Output rows are PADDED to a multiple of 64 by the caller, so all stores are
// unconditional (no exec-mask churn).  bias is always a valid pointer.
// ---------------------------------------------------------------------------
__global__ __launch_bounds__(128)
void gemm_bf16_kernel(const float* __restrict__ A, const int* __restrict__ idx,
                      const __bf16* __restrict__ Wp, const float* __restrict__ bias,
                      float* __restrict__ out, int nclamp) {
    const int lane = threadIdx.x & 31;
    const int wave = threadIdx.x >> 5;
    const int m0   = blockIdx.x * 64 + wave * 16;
    const int lm   = lane & 15;
    const int hi   = lane >> 4;              // 0: K base +0, 1: K base +8

    // ---- A fragments (clamped row keeps loads in-bounds, EXEC stays full) ----
    int row  = m0 + lm;
    int crow = row < nclamp ? row : (nclamp - 1);
    long arow = idx ? (long)idx[crow] : (long)crow;
    const float* ap = A + arow * HID + hi * 8;

    v16bf afrag[4];
#pragma unroll
    for (int kt = 0; kt < 4; ++kt) {
        const float4* p = (const float4*)(ap + kt * 32);
        float4 f0 = p[0];   // k-local 0..3
        float4 f1 = p[1];   // k-local 4..7
        float4 f2 = p[4];   // k-local 16..19
        float4 f3 = p[5];   // k-local 20..23
        v16bf a;
        a[0]  = (__bf16)f0.x; a[1]  = (__bf16)f0.y; a[2]  = (__bf16)f0.z; a[3]  = (__bf16)f0.w;
        a[4]  = (__bf16)f1.x; a[5]  = (__bf16)f1.y; a[6]  = (__bf16)f1.z; a[7]  = (__bf16)f1.w;
        a[8]  = (__bf16)f2.x; a[9]  = (__bf16)f2.y; a[10] = (__bf16)f2.z; a[11] = (__bf16)f2.w;
        a[12] = (__bf16)f3.x; a[13] = (__bf16)f3.y; a[14] = (__bf16)f3.z; a[15] = (__bf16)f3.w;
        afrag[kt] = a;
    }

    const v16bf* bp = (const v16bf*)Wp;      // 32 B per lane-chunk, 32 chunks/tile
    const int mb = m0 + hi * 8;              // C/D layout: VGPR r -> M = r (+8 hi)
#pragma unroll
    for (int nt = 0; nt < 8; ++nt) {
        int col = nt * 16 + lm;
        float bv = bias[col];
        v8f acc;
#pragma unroll
        for (int r = 0; r < 8; ++r) acc[r] = bv;
#pragma unroll
        for (int kt = 0; kt < 4; ++kt) {
            v16bf b = bp[(kt * 8 + nt) * 32 + lane];
            acc = __builtin_amdgcn_wmma_f32_16x16x32_bf16(
                      false, afrag[kt], false, b, (short)0, acc, false, false);
        }
        float* o = out + (long)mb * HID + col;
#pragma unroll
        for (int r = 0; r < 8; ++r) o[(long)r * HID] = acc[r];   // rows padded: no guard
    }
}

// ---------------------------------------------------------------------------
// Edge message pass: acc[dst] += t[src] * w.  One wave per edge, float4/lane.
// ---------------------------------------------------------------------------
__global__ __launch_bounds__(256)
void edge_scatter_kernel(const float* __restrict__ t, const int* __restrict__ ei,
                         const float* __restrict__ ew, float* __restrict__ acc, int E) {
    int gid  = blockIdx.x * 256 + threadIdx.x;
    int e    = gid >> 5;
    int lane = gid & 31;
    if (e >= E) return;
    int src = ei[e];
    int dst = ei[E + e];
    float w = ew[e];
    const float4 v = *(const float4*)(t + (long)src * HID + lane * 4);
    float* d = acc + (long)dst * HID + lane * 4;
    atomicAdd(d + 0, v.x * w);
    atomicAdd(d + 1, v.y * w);
    atomicAdd(d + 2, v.z * w);
    atomicAdd(d + 3, v.w * w);
}

__global__ __launch_bounds__(256)
void bias_relu_kernel(float* __restrict__ h, const float* __restrict__ b, long n) {
    long i = (long)blockIdx.x * 256 + threadIdx.x;
    if (i >= n) return;
    float v = h[i] + b[i & (HID - 1)];
    h[i] = v > 0.0f ? v : 0.0f;
}

__global__ __launch_bounds__(256)
void fill_kernel(float* __restrict__ p, long n, float v) {
    long i = (long)blockIdx.x * 256 + threadIdx.x;
    if (i < n) p[i] = v;
}

__global__ __launch_bounds__(256)
void init_out_kernel(float* __restrict__ out) {
    int g = blockIdx.x, c = threadIdx.x;          // 128 x 256
    out[g * 256 + c] = (c < HID) ? 0.0f : -__builtin_inff();
}

__device__ __forceinline__ void atomic_max_f32(float* addr, float val) {
    if (val >= 0.0f) atomicMax((int*)addr, __float_as_int(val));
    else             atomicMin((unsigned int*)addr, __float_as_uint(val));
}

// Per-graph mean(sum)/max pooling: one wave per node, float4 per lane.
__global__ __launch_bounds__(256)
void pool_kernel(const float* __restrict__ h, const int* __restrict__ batch,
                 float* __restrict__ out, float* __restrict__ cnt, int n) {
    int gid  = blockIdx.x * 256 + threadIdx.x;
    int node = gid >> 5;
    int lane = gid & 31;
    if (node >= n) return;
    int g = batch[node];
    if (lane == 0) atomicAdd(&cnt[g], 1.0f);
    const float4 v = *(const float4*)(h + (long)node * HID + lane * 4);
    float* mean = out + (long)g * 256 + lane * 4;
    float* mx   = mean + HID;
    atomicAdd(mean + 0, v.x); atomicAdd(mean + 1, v.y);
    atomicAdd(mean + 2, v.z); atomicAdd(mean + 3, v.w);
    atomic_max_f32(mx + 0, v.x); atomic_max_f32(mx + 1, v.y);
    atomic_max_f32(mx + 2, v.z); atomic_max_f32(mx + 3, v.w);
}

__global__ __launch_bounds__(128)
void finalize_kernel(float* __restrict__ out, const float* __restrict__ cnt) {
    int g = blockIdx.x, c = threadIdx.x;          // 128 x 128 (mean half only)
    float cc = cnt[g];
    cc = cc > 1.0f ? cc : 1.0f;
    out[g * 256 + c] /= cc;
}

// ---------------------------------------------------------------------------
extern "C" void kernel_launch(void* const* d_in, const int* in_sizes, int n_in,
                              void* d_out, int out_size, void* d_ws, size_t ws_size,
                              hipStream_t stream) {
    (void)n_in; (void)out_size; (void)ws_size;
    const int*   x    = (const int*)d_in[0];
    const int*   ei   = (const int*)d_in[1];   // [2, E] flat: src then dst
    const float* ew   = (const float*)d_in[2];
    const int*   bat  = (const int*)d_in[3];
    const float* emb  = (const float*)d_in[4];
    const float* W1   = (const float*)d_in[5];
    const float* b1   = (const float*)d_in[6];
    const float* W2   = (const float*)d_in[7];
    const float* b2   = (const float*)d_in[8];
    const float* Wfc  = (const float*)d_in[9];
    const float* bfc  = (const float*)d_in[10];
    const int N    = in_sizes[0];
    const int Npad = (N + 63) & ~63;           // GEMM writes padded rows -> no store guards
    const int E    = in_sizes[2];
    float* outp = (float*)d_out;

    char* ws = (char*)d_ws;
    __bf16* Wp1  = (__bf16*)(ws);
    __bf16* Wp2  = (__bf16*)(ws + 32768);
    __bf16* Wpf  = (__bf16*)(ws + 65536);
    float*  cnt  = (float*)(ws + 98304);               // 128 floats
    float*  zb   = (float*)(ws + 98304 + 512);         // 128-float zero bias
    float*  bufA = (float*)(ws + 98304 + 1024);
    float*  bufB = (float*)(ws + 98304 + 1024 + (size_t)Npad * HID * sizeof(float));

    const long NH = (long)N * HID;
    const int  nhBlocks   = (int)((NH + 255) / 256);
    const int  gemmBlocks = Npad / 64;
    const int  edgeBlocks = (int)(((long)E * 32 + 255) / 256);
    const int  poolBlocks = (int)(((long)N * 32 + 255) / 256);

    // weight packing (bf16 B-fragment layout) + zero/init
    pack_w_kernel<<<64, 256, 0, stream>>>(W1, Wp1);
    pack_w_kernel<<<64, 256, 0, stream>>>(W2, Wp2);
    pack_w_kernel<<<64, 256, 0, stream>>>(Wfc, Wpf);
    fill_kernel<<<nhBlocks, 256, 0, stream>>>(bufB, NH, 0.0f);
    fill_kernel<<<1, 256, 0, stream>>>(cnt, GRAPHS + HID, 0.0f);   // cnt + zero-bias
    init_out_kernel<<<GRAPHS, 256, 0, stream>>>(outp);

    // layer 1: fused embedding-gather GEMM, scatter, bias+relu
    gemm_bf16_kernel<<<gemmBlocks, 128, 0, stream>>>(emb, x, Wp1, zb, bufA, N);
    edge_scatter_kernel<<<edgeBlocks, 256, 0, stream>>>(bufA, ei, ew, bufB, E);
    bias_relu_kernel<<<nhBlocks, 256, 0, stream>>>(bufB, b1, NH);

    // layer 2
    gemm_bf16_kernel<<<gemmBlocks, 128, 0, stream>>>(bufB, nullptr, Wp2, zb, bufA, N);
    fill_kernel<<<nhBlocks, 256, 0, stream>>>(bufB, NH, 0.0f);
    edge_scatter_kernel<<<edgeBlocks, 256, 0, stream>>>(bufA, ei, ew, bufB, E);
    bias_relu_kernel<<<nhBlocks, 256, 0, stream>>>(bufB, b2, NH);

    // fc (+bias) and pooling
    gemm_bf16_kernel<<<gemmBlocks, 128, 0, stream>>>(bufB, nullptr, Wpf, bfc, bufA, N);
    pool_kernel<<<poolBlocks, 256, 0, stream>>>(bufA, bat, outp, cnt, N);
    finalize_kernel<<<GRAPHS, 128, 0, stream>>>(outp, cnt);
}
